// GPT2DecoderLayer_32822140076235
// MI455X (gfx1250) — matrix operand order repaired and checked by reading, not compiled, not run on hardware
//
#include <hip/hip_runtime.h>
#include <hip/hip_bf16.h>

// ---------------------------------------------------------------------------
// GPT-2 decoder layer forward for gfx1250 (MI455X), wave32 + WMMA f16 +
// async global->LDS staging (ASYNCcnt pipeline, double-buffered LDS).
// Outputs concatenated in d_out: [ y (BS*D fp32) | attn_p (B*H*S*S fp32) ]
// ---------------------------------------------------------------------------

typedef __attribute__((ext_vector_type(16))) _Float16 v16h;
typedef __attribute__((ext_vector_type(8)))  float    v8f;

#define LDS_PAD 40   // halves per LDS row (32 data + 8 pad)

union AFragU { v16h v; uint4 q[2]; };

__device__ __forceinline__ unsigned lds_off(const void* p)
{
    // generic pointers to LDS: low 32 bits are the LDS byte offset (ISA 10.2)
    return (unsigned)(size_t)p;
}

// EPI: 0 = store f16
//      1 = scale 1/sqrt(64) + causal mask, store fp32   (attention scores)
//      2 = + bias, store fp32
//      3 = + bias, exact GELU, store f16
template <typename TC, bool B_KMAJOR, int EPI>
__global__ __launch_bounds__(256)
void gemm_wmma(const _Float16* __restrict__ A, const _Float16* __restrict__ B,
               TC* __restrict__ C, const float* __restrict__ bias,
               int M, int N, int K, int lda, int ldb, int ldc,
               long sAb, long sAh, long sBb, long sBh, long sCb, long sCh,
               int Hz)
{
    __shared__ _Float16 ldsA[2][128 * LDS_PAD];
    __shared__ _Float16 ldsB[2][64  * LDS_PAD];

    const int tid  = threadIdx.x;
    const int lane = tid & 31;
    const int wid  = tid >> 5;
    const int wm   = wid >> 1;     // 0..3  (wave row, 32 rows each)
    const int wn   = wid & 1;      // 0..1  (wave col, 32 cols each)

    const int z  = blockIdx.z;
    const int bz = z / Hz, hz = z % Hz;
    A += (size_t)bz * sAb + (size_t)hz * sAh;
    B += (size_t)bz * sBb + (size_t)hz * sBh;
    C += (size_t)bz * sCb + (size_t)hz * sCh;

    const int tileM = blockIdx.x * 128;
    const int tileN = blockIdx.y * 64;

    const int halfSel   = (lane < 16) ? 0 : 1;
    const int rowInHalf = lane & 15;

    // ---- async staging of one K-step (32) into LDS buffer `buf` ------------
    auto stage = [&](int buf, int k0) {
        {   // A tile 128x32: 16 halves (32B) per thread, 2 async b128 per wave-op
            const int r  = tid >> 1;
            const int kb = (tid & 1) * 16;
            const _Float16* src = A + (size_t)(tileM + r) * lda + k0 + kb;
            unsigned dofs = lds_off(&ldsA[buf][r * LDS_PAD + kb]);
            unsigned long long s64 = (unsigned long long)(size_t)src;
            asm volatile(
                "global_load_async_to_lds_b128 %0, %1, off\n\t"
                "global_load_async_to_lds_b128 %0, %1, off offset:16"
                :: "v"(dofs), "v"(s64) : "memory");
        }
        if constexpr (B_KMAJOR) {   // memory B[n*ldb + k]: 8 halves per thread
            const int n  = tid >> 2;
            const int kb = (tid & 3) * 8;
            const _Float16* src = B + (size_t)(tileN + n) * ldb + k0 + kb;
            unsigned dofs = lds_off(&ldsB[buf][n * LDS_PAD + kb]);
            asm volatile(
                "global_load_async_to_lds_b128 %0, %1, off"
                :: "v"(dofs), "v"((unsigned long long)(size_t)src) : "memory");
        } else {                    // memory B[k*ldb + n]: transpose via ds stores
            const int kk = tid >> 3;
            const int nb = (tid & 7) * 8;
            const _Float16* src = B + (size_t)(k0 + kk) * ldb + tileN + nb;
#pragma unroll
            for (int i = 0; i < 8; ++i)
                ldsB[buf][(nb + i) * LDS_PAD + kk] = src[i];
        }
    };

    v8f acc[2][2] = {};
    const int nk = K >> 5;

    stage(0, 0);                         // prologue
    for (int t = 0; t < nk; ++t) {
        asm volatile("s_wait_asynccnt 0x0" ::: "memory");
        __syncthreads();                 // buffer t fully resident in LDS
        if (t + 1 < nk) stage((t + 1) & 1, (t + 1) * 32);  // overlap next tile

        const _Float16* bufA = ldsA[t & 1];
        const _Float16* bufB = ldsB[t & 1];

        // A 16x32 frag: lanes 0-15 own K=0..7,16..23 ; lanes 16-31 K=8..15,24..31
        AFragU a[2], b[2];
#pragma unroll
        for (int i = 0; i < 2; ++i) {
            const _Float16* ap =
                &bufA[(wm * 32 + i * 16 + rowInHalf) * LDS_PAD + halfSel * 8];
            a[i].q[0] = *(const uint4*)(ap);
            a[i].q[1] = *(const uint4*)(ap + 16);
        }
        // B 32x16 frag: lanes 0-15 own K=0..7,16..23 ; lanes 16-31 K=8..15,24..31
#pragma unroll
        for (int j = 0; j < 2; ++j) {
            const _Float16* bp =
                &bufB[(wn * 32 + j * 16 + rowInHalf) * LDS_PAD + halfSel * 8];
            b[j].q[0] = *(const uint4*)(bp);
            b[j].q[1] = *(const uint4*)(bp + 16);
        }
#pragma unroll
        for (int i = 0; i < 2; ++i)
#pragma unroll
            for (int j = 0; j < 2; ++j)
                acc[i][j] = __builtin_amdgcn_wmma_f32_16x16x32_f16(
                    false, a[i].v, false, b[j].v, (short)0, acc[i][j],
                    false, false);
    }

    // ---- epilogue: C/D layout VGPR r -> M = r (+8 for lanes 16-31) ---------
#pragma unroll
    for (int i = 0; i < 2; ++i) {
#pragma unroll
        for (int j = 0; j < 2; ++j) {
            const int colg = tileN + wn * 32 + j * 16 + rowInHalf;
            float bv = 0.0f;
            if constexpr (EPI == 2 || EPI == 3) bv = bias[colg];
#pragma unroll
            for (int r = 0; r < 8; ++r) {
                const int rowg = tileM + wm * 32 + i * 16 + r + halfSel * 8;
                float v = acc[i][j][r];
                if constexpr (EPI == 0) {
                    C[(size_t)rowg * ldc + colg] = (TC)v;
                } else if constexpr (EPI == 1) {
                    v *= 0.125f;                       // 1/sqrt(DH=64)
                    if (colg > rowg) v = -1e9f;        // causal mask
                    C[(size_t)rowg * ldc + colg] = v;
                } else if constexpr (EPI == 2) {
                    C[(size_t)rowg * ldc + colg] = v + bv;
                } else {                               // bias + exact GELU -> f16
                    float t = v + bv;
                    float g = 0.5f * t *
                              (1.0f + erff(t * 0.70710678118654752f));
                    C[(size_t)rowg * ldc + colg] = (TC)g;
                }
            }
        }
    }
}

// ---- fp32 -> f16 pre-conversion (weights / activations), 4 elems/thread ----
__global__ __launch_bounds__(256)
void cvt_f32_f16(const float* __restrict__ s, _Float16* __restrict__ d, int n4)
{
    int i = blockIdx.x * 256 + threadIdx.x;
    if (i < n4) {
        float4 f = ((const float4*)s)[i];
        union { _Float16 h[4]; uint2 u; } t;
        t.h[0] = (_Float16)f.x; t.h[1] = (_Float16)f.y;
        t.h[2] = (_Float16)f.z; t.h[3] = (_Float16)f.w;
        ((uint2*)d)[i] = t.u;
    }
}

// ---- row softmax over S=2048, in-place fp32 + f16 copy ---------------------
__global__ __launch_bounds__(256)
void softmax_rows(float* __restrict__ P, _Float16* __restrict__ P16)
{
    __shared__ float red[256];
    const size_t row = blockIdx.x;
    float*    p  = P   + row * 2048;
    _Float16* ph = P16 + row * 2048;
    const int tid = threadIdx.x;

    float vals[8];
    float mx = -1e30f;
#pragma unroll
    for (int i = 0; i < 8; ++i) {
        vals[i] = p[tid + i * 256];
        mx = fmaxf(mx, vals[i]);
    }
    red[tid] = mx; __syncthreads();
    for (int s = 128; s > 0; s >>= 1) {
        if (tid < s) red[tid] = fmaxf(red[tid], red[tid + s]);
        __syncthreads();
    }
    mx = red[0]; __syncthreads();

    float sum = 0.0f;
#pragma unroll
    for (int i = 0; i < 8; ++i) { vals[i] = __expf(vals[i] - mx); sum += vals[i]; }
    red[tid] = sum; __syncthreads();
    for (int s = 128; s > 0; s >>= 1) {
        if (tid < s) red[tid] += red[tid + s];
        __syncthreads();
    }
    const float inv = 1.0f / red[0];
#pragma unroll
    for (int i = 0; i < 8; ++i) {
        float o = vals[i] * inv;
        p[tid + i * 256]  = o;
        ph[tid + i * 256] = (_Float16)o;
    }
}

// ---- residual add + LayerNorm over D=768, optional f16 copy ----------------
__global__ __launch_bounds__(256)
void add_ln(const float* __restrict__ X, const float* __restrict__ Yd,
            const float* __restrict__ g, const float* __restrict__ bt,
            float* __restrict__ out32, _Float16* __restrict__ out16)
{
    __shared__ float red[256];
    __shared__ float red2[256];
    const size_t row = blockIdx.x;
    const float* x = X  + row * 768;
    const float* y = Yd + row * 768;
    const int tid = threadIdx.x;

    float v[3], s = 0.0f, s2 = 0.0f;
#pragma unroll
    for (int i = 0; i < 3; ++i) {
        float t = x[tid + i * 256] + y[tid + i * 256];
        v[i] = t; s += t; s2 += t * t;
    }
    red[tid] = s; red2[tid] = s2; __syncthreads();
    for (int st = 128; st > 0; st >>= 1) {
        if (tid < st) { red[tid] += red[tid + st]; red2[tid] += red2[tid + st]; }
        __syncthreads();
    }
    const float mu  = red[0] * (1.0f / 768.0f);
    const float var = red2[0] * (1.0f / 768.0f) - mu * mu;
    const float rs  = rsqrtf(var + 1e-5f);
#pragma unroll
    for (int i = 0; i < 3; ++i) {
        const int c = tid + i * 256;
        float o = (v[i] - mu) * rs * g[c] + bt[c];
        out32[row * 768 + c] = o;
        if (out16) out16[row * 768 + c] = (_Float16)o;
    }
}

extern "C" void kernel_launch(void* const* d_in, const int* in_sizes, int n_in,
                              void* d_out, int out_size, void* d_ws, size_t ws_size,
                              hipStream_t stream)
{
    (void)in_sizes; (void)n_in; (void)out_size; (void)ws_size;
    // inputs: x, attn_mask, Wq, Wk, Wv, Wo, bo, g1, b1, W1, bb1, W2, bb2, g2, b2
    const float* x   = (const float*)d_in[0];
    const float* Wq  = (const float*)d_in[2];
    const float* Wk  = (const float*)d_in[3];
    const float* Wv  = (const float*)d_in[4];
    const float* Wo  = (const float*)d_in[5];
    const float* bo  = (const float*)d_in[6];
    const float* g1  = (const float*)d_in[7];
    const float* b1  = (const float*)d_in[8];
    const float* W1  = (const float*)d_in[9];
    const float* bb1 = (const float*)d_in[10];
    const float* W2  = (const float*)d_in[11];
    const float* bb2 = (const float*)d_in[12];
    const float* g2  = (const float*)d_in[13];
    const float* b2  = (const float*)d_in[14];

    constexpr int S = 2048, H = 12, D = 768, DF = 3072, BS = 4096;
    constexpr long SS = (long)S * S;

    float* y     = (float*)d_out;
    float* attnP = y + (size_t)BS * D;        // scores -> softmax in place

    // ---- workspace layout (all offsets 256B-aligned) -----------------------
    char* ws = (char*)d_ws;
    _Float16* X16  = (_Float16*)(ws + 0);          //  6291456 B
    _Float16* Wq16 = (_Float16*)(ws + 6291456);    //  1179648 B
    _Float16* Wk16 = (_Float16*)(ws + 7471104);
    _Float16* Wv16 = (_Float16*)(ws + 8650752);
    _Float16* Wo16 = (_Float16*)(ws + 9830400);
    _Float16* W116 = (_Float16*)(ws + 11010048);   //  4718592 B
    _Float16* W216 = (_Float16*)(ws + 15728640);   //  4718592 B
    _Float16* Q16  = (_Float16*)(ws + 20447232);   //  6291456 B
    _Float16* K16  = (_Float16*)(ws + 26738688);
    _Float16* V16  = (_Float16*)(ws + 33030144);
    _Float16* AV16 = (_Float16*)(ws + 39321600);
    _Float16* H16  = (_Float16*)(ws + 45613056);
    float*    H32  = (float*)   (ws + 51904512);   // 12582912 B
    char* big = ws + 64487424;                     // 201.3 MB region
    _Float16* P16  = (_Float16*)big;
    float*    AO32 = (float*)big;                  // alias (after P16 dead)
    _Float16* F116 = (_Float16*)(big + 12582912);
    float*    F232 = (float*)   (big + 12582912 + 25165824);

    const dim3 blk(256);

    // 0) one-time fp32 -> f16 conversion of activations + weights
    cvt_f32_f16<<<dim3(3072), blk, 0, stream>>>(x,  X16,  786432);   // 4096*768/4
    cvt_f32_f16<<<dim3(576),  blk, 0, stream>>>(Wq, Wq16, 147456);   // 768*768/4
    cvt_f32_f16<<<dim3(576),  blk, 0, stream>>>(Wk, Wk16, 147456);
    cvt_f32_f16<<<dim3(576),  blk, 0, stream>>>(Wv, Wv16, 147456);
    cvt_f32_f16<<<dim3(576),  blk, 0, stream>>>(Wo, Wo16, 147456);
    cvt_f32_f16<<<dim3(2304), blk, 0, stream>>>(W1, W116, 589824);   // 3072*768/4
    cvt_f32_f16<<<dim3(2304), blk, 0, stream>>>(W2, W216, 589824);

    // 1) QKV projections: [4096x768] = x @ W^T, store f16
    gemm_wmma<_Float16, true, 0><<<dim3(32, 12, 1), blk, 0, stream>>>(
        X16, Wq16, Q16, nullptr, BS, D, D, D, D, D, 0, 0, 0, 0, 0, 0, 1);
    gemm_wmma<_Float16, true, 0><<<dim3(32, 12, 1), blk, 0, stream>>>(
        X16, Wk16, K16, nullptr, BS, D, D, D, D, D, 0, 0, 0, 0, 0, 0, 1);
    gemm_wmma<_Float16, true, 0><<<dim3(32, 12, 1), blk, 0, stream>>>(
        X16, Wv16, V16, nullptr, BS, D, D, D, D, D, 0, 0, 0, 0, 0, 0, 1);

    // 2) scores[b,h] = Q K^T / 8 with causal mask -> attn_p region (fp32)
    gemm_wmma<float, true, 1><<<dim3(16, 32, 24), blk, 0, stream>>>(
        Q16, K16, attnP, nullptr, S, S, 64, D, D, S,
        (long)S * D, 64, (long)S * D, 64, (long)H * SS, SS, H);

    // 3) softmax rows in place + f16 copy for P@V
    softmax_rows<<<dim3(2 * H * S), blk, 0, stream>>>(attnP, P16);

    // 4) attn_v[b,h] = P @ V   (B is [k][n] in memory: B_KMAJOR=false)
    gemm_wmma<_Float16, false, 0><<<dim3(16, 1, 24), blk, 0, stream>>>(
        P16, V16, AV16, nullptr, S, 64, S, S, D, D,
        (long)H * SS, SS, (long)S * D, 64, (long)S * D, 64, H);

    // 5) attn_out = attn_v @ Wo^T + bo (fp32)
    gemm_wmma<float, true, 2><<<dim3(32, 12, 1), blk, 0, stream>>>(
        AV16, Wo16, AO32, bo, BS, D, D, D, D, D, 0, 0, 0, 0, 0, 0, 1);

    // 6) h = LN1(x + attn_out) -> fp32 + f16
    add_ln<<<dim3(BS), blk, 0, stream>>>(x, AO32, g1, b1, H32, H16);

    // 7) f1 = GELU(h @ W1^T + bb1) -> f16  [4096 x 3072]
    gemm_wmma<_Float16, true, 3><<<dim3(32, 48, 1), blk, 0, stream>>>(
        H16, W116, F116, bb1, BS, DF, D, D, D, DF, 0, 0, 0, 0, 0, 0, 1);

    // 8) f2 = f1 @ W2^T + bb2 -> fp32  [4096 x 768]
    gemm_wmma<float, true, 2><<<dim3(32, 12, 1), blk, 0, stream>>>(
        F116, W216, F232, bb2, BS, D, DF, DF, DF, D, 0, 0, 0, 0, 0, 0, 1);

    // 9) y = LN2(h + f2) -> d_out
    add_ln<<<dim3(BS), blk, 0, stream>>>(H32, F232, g2, b2, y, nullptr);
}
